// FPModule_26336739459588
// MI455X (gfx1250) — matrix operand order repaired and checked by reading, not compiled, not run on hardware
//
#include <hip/hip_runtime.h>
#include <hip/hip_bf16.h>

// ---------------------------------------------------------------------------
// FPModule (PointNet++ FP layer): knn_interpolate(k=3) + concat + 2-layer MLP
// MI455X / gfx1250, wave32, WMMA 16x16x32 bf16 (f32 accumulate).
//
// Shapes (fixed by reference):
//   B=16, n_coarse=1024, n_fine=4096  -> N_c=16384, N_f=65536
//   C_in=256, C_skip=128, C_mid=256, C_out=256
// Output tuple laid out flat in d_out:
//   [0, 65536*256)                      : out (f32)
//   [65536*256, +65536*3)               : pos_skip copy (f32)
//   then 65536 int64 batch_skip (raw bits)
// ---------------------------------------------------------------------------

typedef __attribute__((ext_vector_type(16))) __bf16         v16bf;
typedef __attribute__((ext_vector_type(8)))  float          v8f;
typedef __attribute__((ext_vector_type(8)))  unsigned short ush8;
typedef __attribute__((ext_vector_type(16))) unsigned short ush16;

#define NCOARSE   1024
#define NFINE_PER 4096
#define CIN       256
#define CSKIP     128
#define KDIM1     384     // CIN + CSKIP
#define KDIM2     256
#define NOUT      256
#define KT1       12      // KDIM1 / 32
#define KT2       8       // KDIM2 / 32
#define ROWS      32      // fine rows per workgroup
#define SA        392     // hA row stride (bf16 elems), 784B = 16B-aligned
#define SB        264     // hB row stride (bf16 elems), 528B = 16B-aligned
#define OUT_ELEMS (65536 * 256)
#define POS_ELEMS (65536 * 3)

__device__ __forceinline__ unsigned short f2bf(float f) {
  unsigned int u = __float_as_uint(f);
  unsigned int r = u + 0x7FFFu + ((u >> 16) & 1u);   // round-to-nearest-even
  return (unsigned short)(r >> 16);
}

__device__ __forceinline__ v16bf make_frag(ush8 lo, ush8 hi) {
  ush16 u = __builtin_shufflevector(lo, hi, 0, 1, 2, 3, 4, 5, 6, 7,
                                            8, 9, 10, 11, 12, 13, 14, 15);
  union { ush16 u; v16bf b; } cvt;
  cvt.u = u;
  return cvt.b;
}

// ---------------------------------------------------------------------------
// Pack f32 row-major W[K][256] into bf16 B-fragment order:
//   Wp[((nt*KT + kt)*32 + lane)*16 + e] = bf16( W[(kt*32 + (lane>>4)*8 + (e&7) + (e>>3)*16)*256
//                                                + nt*16 + (lane&15)] )
// so each lane's V_WMMA B operand is one contiguous 32-byte global load.
// ---------------------------------------------------------------------------
__global__ void pack_weights_kernel(const float* __restrict__ W,
                                    unsigned short* __restrict__ Wp,
                                    int KT) {
  int gid   = blockIdx.x * 256 + threadIdx.x;
  int total = 16 * KT * 512;
  if (gid >= total) return;
  int e    = gid & 15;
  int lane = (gid >> 4) & 31;
  int q    = gid >> 9;
  int kt   = q % KT;
  int nt   = q / KT;
  int n = nt * 16 + (lane & 15);
  int k = kt * 32 + (lane >> 4) * 8 + (e & 7) + (e >> 3) * 16;
  Wp[gid] = f2bf(W[(size_t)k * 256 + n]);
}

// ---------------------------------------------------------------------------
// Fused KNN + interpolate + MLP. 128 threads (4 wave32) per block,
// 32 fine points per block. 2048 blocks.
// ---------------------------------------------------------------------------
__global__ __launch_bounds__(128) void fp_fused_kernel(
    const float* __restrict__ x,         // [16384, 256]
    const float* __restrict__ pos,       // [16384, 3]
    const float* __restrict__ x_skip,    // [65536, 128]
    const float* __restrict__ pos_skip,  // [65536, 3]
    const unsigned short* __restrict__ Wp1,  // packed bf16, 16*12*512
    const float* __restrict__ b1,        // [256]
    const unsigned short* __restrict__ Wp2,  // packed bf16, 16*8*512
    const float* __restrict__ b2,        // [256]
    float* __restrict__ out)             // [65536, 256]
{
  __shared__ __align__(16) unsigned short sh_hA[ROWS * SA];  // 25088 B
  __shared__ __align__(16) unsigned short sh_hB[ROWS * SB];  // 16896 B
  __shared__ float sh_cpos[NCOARSE * 3];                     // 12288 B
  __shared__ float sh_sd[4 * 96];
  __shared__ int   sh_si[4 * 96];
  __shared__ float sh_w[ROWS * 3];
  __shared__ int   sh_idx[ROWS * 3];

  const int tid  = threadIdx.x;
  const int lane = tid & 31;
  const int wv   = tid >> 5;

  const int f0    = blockIdx.x * ROWS;          // first fine row
  const int cloud = f0 >> 12;                   // /4096
  const int cbase = cloud << 10;                // *1024

  // ---- stage coarse positions of this cloud into LDS -----------------------
  for (int i = tid; i < NCOARSE * 3; i += 128)
    sh_cpos[i] = pos[(size_t)cbase * 3 + i];
  __syncthreads();

  // ---- KNN: each wave handles 8 points; lanes scan coarse set --------------
  for (int it = 0; it < 8; ++it) {
    const int pl = wv * 8 + it;
    const int p  = f0 + pl;
    const float qx = pos_skip[(size_t)p * 3 + 0];
    const float qy = pos_skip[(size_t)p * 3 + 1];
    const float qz = pos_skip[(size_t)p * 3 + 2];

    float d0 = 3.4e38f, d1 = 3.4e38f, d2v = 3.4e38f;
    int   i0 = 0, i1 = 0, i2 = 0;
    for (int j = lane; j < NCOARSE; j += 32) {
      float dx = qx - sh_cpos[3 * j + 0];
      float dy = qy - sh_cpos[3 * j + 1];
      float dz = qz - sh_cpos[3 * j + 2];
      float d = dx * dx + dy * dy + dz * dz;
      if (d < d2v) {
        if (d < d1) {
          d2v = d1; i2 = i1;
          if (d < d0) { d1 = d0; i1 = i0; d0 = d; i0 = j; }
          else        { d1 = d;  i1 = j; }
        } else { d2v = d; i2 = j; }
      }
    }
    sh_sd[wv * 96 + lane * 3 + 0] = d0;  sh_si[wv * 96 + lane * 3 + 0] = i0;
    sh_sd[wv * 96 + lane * 3 + 1] = d1;  sh_si[wv * 96 + lane * 3 + 1] = i1;
    sh_sd[wv * 96 + lane * 3 + 2] = d2v; sh_si[wv * 96 + lane * 3 + 2] = i2;
    __syncthreads();
    if (lane == 0) {
      float e0 = 3.4e38f, e1 = 3.4e38f, e2 = 3.4e38f;
      int   j0 = 0, j1 = 0, j2 = 0;
      for (int e = 0; e < 96; ++e) {
        float d = sh_sd[wv * 96 + e];
        int  id = sh_si[wv * 96 + e];
        if (d < e2) {
          if (d < e1) {
            e2 = e1; j2 = j1;
            if (d < e0) { e1 = e0; j1 = j0; e0 = d; j0 = id; }
            else        { e1 = d;  j1 = id; }
          } else { e2 = d; j2 = id; }
        }
      }
      float w0 = 1.0f / fmaxf(e0, 1e-16f);
      float w1 = 1.0f / fmaxf(e1, 1e-16f);
      float w2 = 1.0f / fmaxf(e2, 1e-16f);
      float inv = 1.0f / (w0 + w1 + w2);
      sh_w[pl * 3 + 0] = w0 * inv;  sh_idx[pl * 3 + 0] = cbase + j0;
      sh_w[pl * 3 + 1] = w1 * inv;  sh_idx[pl * 3 + 1] = cbase + j1;
      sh_w[pl * 3 + 2] = w2 * inv;  sh_idx[pl * 3 + 2] = cbase + j2;
    }
    __syncthreads();
  }

  // ---- interpolate features + skip concat into LDS (bf16) ------------------
  for (int pl = 0; pl < ROWS; ++pl) {
    const int   r0 = sh_idx[pl * 3 + 0], r1 = sh_idx[pl * 3 + 1], r2 = sh_idx[pl * 3 + 2];
    const float w0 = sh_w[pl * 3 + 0],  w1 = sh_w[pl * 3 + 1],  w2 = sh_w[pl * 3 + 2];
    for (int c = tid; c < CIN; c += 128) {
      float y = w0 * x[(size_t)r0 * CIN + c]
              + w1 * x[(size_t)r1 * CIN + c]
              + w2 * x[(size_t)r2 * CIN + c];
      sh_hA[pl * SA + c] = f2bf(y);
    }
    sh_hA[pl * SA + CIN + tid] = f2bf(x_skip[(size_t)(f0 + pl) * CSKIP + tid]);
  }
  __syncthreads();

  // ---- GEMM1: relu(hA[32x384] @ W1 + b1) -> hB (bf16 in LDS) ---------------
  // wave -> rows 16*(wv>>1)..+15, cols ((wv&1)*8 + chunk*4)*16 ..
  const int rowbase = (wv >> 1) * 16;
  const int half    = lane >> 4;
  const int ln15    = lane & 15;
  const unsigned short* arow1 = sh_hA + (rowbase + ln15) * SA;

  for (int chunk = 0; chunk < 2; ++chunk) {
    const int nts = (wv & 1) * 8 + chunk * 4;
    v8f acc[4] = { {0,0,0,0,0,0,0,0}, {0,0,0,0,0,0,0,0},
                   {0,0,0,0,0,0,0,0}, {0,0,0,0,0,0,0,0} };
    for (int kt = 0; kt < KT1; ++kt) {
      const int k0 = kt * 32 + half * 8;
      ush8 alo = *(const ush8*)(arow1 + k0);
      ush8 ahi = *(const ush8*)(arow1 + k0 + 16);
      v16bf a = make_frag(alo, ahi);
#pragma unroll
      for (int t = 0; t < 4; ++t) {
        const unsigned short* bp =
            Wp1 + ((size_t)((nts + t) * KT1 + kt) * 512) + lane * 16;
        ush8 blo = *(const ush8*)(bp);
        ush8 bhi = *(const ush8*)(bp + 8);
        v16bf b = make_frag(blo, bhi);
        acc[t] = __builtin_amdgcn_wmma_f32_16x16x32_bf16(
            false, a, false, b, (short)0, acc[t], false, false);
      }
    }
#pragma unroll
    for (int t = 0; t < 4; ++t) {
      const int n = (nts + t) * 16 + ln15;
      const float bias = b1[n];
#pragma unroll
      for (int r = 0; r < 8; ++r) {
        float v = acc[t][r] + bias;
        v = v > 0.0f ? v : 0.0f;
        sh_hB[(rowbase + r + 8 * half) * SB + n] = f2bf(v);
      }
    }
  }
  __syncthreads();

  // ---- GEMM2: relu(hB[32x256] @ W2 + b2) -> out (f32, global) --------------
  const unsigned short* arow2 = sh_hB + (rowbase + ln15) * SB;
  for (int chunk = 0; chunk < 2; ++chunk) {
    const int nts = (wv & 1) * 8 + chunk * 4;
    v8f acc[4] = { {0,0,0,0,0,0,0,0}, {0,0,0,0,0,0,0,0},
                   {0,0,0,0,0,0,0,0}, {0,0,0,0,0,0,0,0} };
    for (int kt = 0; kt < KT2; ++kt) {
      const int k0 = kt * 32 + half * 8;
      ush8 alo = *(const ush8*)(arow2 + k0);
      ush8 ahi = *(const ush8*)(arow2 + k0 + 16);
      v16bf a = make_frag(alo, ahi);
#pragma unroll
      for (int t = 0; t < 4; ++t) {
        const unsigned short* bp =
            Wp2 + ((size_t)((nts + t) * KT2 + kt) * 512) + lane * 16;
        ush8 blo = *(const ush8*)(bp);
        ush8 bhi = *(const ush8*)(bp + 8);
        v16bf b = make_frag(blo, bhi);
        acc[t] = __builtin_amdgcn_wmma_f32_16x16x32_bf16(
            false, a, false, b, (short)0, acc[t], false, false);
      }
    }
#pragma unroll
    for (int t = 0; t < 4; ++t) {
      const int n = (nts + t) * 16 + ln15;
      const float bias = b2[n];
#pragma unroll
      for (int r = 0; r < 8; ++r) {
        float v = acc[t][r] + bias;
        v = v > 0.0f ? v : 0.0f;
        out[(size_t)(f0 + rowbase + r + 8 * half) * NOUT + n] = v;
      }
    }
  }
}

// ---------------------------------------------------------------------------
// Copy tuple outputs 2 (pos_skip) and 3 (batch_skip raw int64 bits) to tail.
// ---------------------------------------------------------------------------
__global__ void tail_copy_kernel(const float* __restrict__ pos_skip,
                                 const unsigned long long* __restrict__ batch_skip,
                                 float* __restrict__ out) {
  int i = blockIdx.x * 256 + threadIdx.x;
  if (i < POS_ELEMS) out[OUT_ELEMS + i] = pos_skip[i];
  if (i < 65536) {
    unsigned long long* tail =
        (unsigned long long*)(out + OUT_ELEMS + POS_ELEMS);
    tail[i] = batch_skip[i];
  }
}

extern "C" void kernel_launch(void* const* d_in, const int* in_sizes, int n_in,
                              void* d_out, int out_size, void* d_ws, size_t ws_size,
                              hipStream_t stream) {
  const float* x        = (const float*)d_in[0];
  const float* pos      = (const float*)d_in[1];
  const float* x_skip   = (const float*)d_in[4];
  const float* pos_skip = (const float*)d_in[5];
  const unsigned long long* batch_skip = (const unsigned long long*)d_in[6];
  const float* W1 = (const float*)d_in[8];
  const float* b1 = (const float*)d_in[9];
  const float* W2 = (const float*)d_in[10];
  const float* b2 = (const float*)d_in[11];
  float* out = (float*)d_out;

  unsigned short* Wp1 = (unsigned short*)d_ws;      // 16*12*512 = 98304 bf16
  unsigned short* Wp2 = Wp1 + 16 * KT1 * 512;       // 16*8*512  = 65536 bf16

  pack_weights_kernel<<<(16 * KT1 * 512 + 255) / 256, 256, 0, stream>>>(W1, Wp1, KT1);
  pack_weights_kernel<<<(16 * KT2 * 512 + 255) / 256, 256, 0, stream>>>(W2, Wp2, KT2);

  fp_fused_kernel<<<65536 / ROWS, 128, 0, stream>>>(
      x, pos, x_skip, pos_skip, Wp1, b1, Wp2, b2, out);

  tail_copy_kernel<<<(POS_ELEMS + 255) / 256, 256, 0, stream>>>(
      pos_skip, batch_skip, out);
}
